// ProbSparseMultiheadAttention_2020_75170517615179
// MI455X (gfx1250) — matrix operand order repaired and checked by reading, not compiled, not run on hardware
//
#include <hip/hip_runtime.h>
#include <hip/hip_bf16.h>

// ---------------------------------------------------------------------------
// ProbSparse multi-head attention for MI455X (gfx1250, wave32, WMMA bf16)
// B=2, L=2048, D=1024, H=16, DK=64, U=40 (padded to 48 rows internally)
// ---------------------------------------------------------------------------

#define B_  2
#define L_  2048
#define D_  1024
#define H_  16
#define DK_ 64
#define U_  40
#define UP_ 48          // U padded to 3 WMMA row-tiles

typedef __bf16 v16bf __attribute__((ext_vector_type(16)));
typedef __bf16 v2bf  __attribute__((ext_vector_type(2)));
typedef float  v8f   __attribute__((ext_vector_type(8)));
typedef float  v2f   __attribute__((ext_vector_type(2)));

union AF {              // one 16x32 bf16 WMMA fragment (8 VGPRs per lane)
    v16bf v;
    unsigned int u[8];
    unsigned short s[16];
};

#define WMMA_BF16(A, Bm, C) \
    __builtin_amdgcn_wmma_f32_16x16x32_bf16(false, (A), false, (Bm), (short)0, (C), false, false)

// hardware f32 -> bf16 conversions (v_cvt_pk_bf16_f32 on gfx1250)
static __device__ __forceinline__ unsigned short f2bf(float x) {
    __bf16 h = (__bf16)x;
    return __builtin_bit_cast(unsigned short, h);
}
static __device__ __forceinline__ unsigned int pack2bf(float lo, float hi) {
    v2f f; f.x = lo; f.y = hi;
    v2bf h = __builtin_convertvector(f, v2bf);
    return __builtin_bit_cast(unsigned int, h);
}

// async-copy helpers (gfx1250 VGLOBAL async-to-LDS; tracked by ASYNCcnt)
static __device__ __forceinline__ void async_b128(const void* gaddr, const void* lds) {
    asm volatile("global_load_async_to_lds_b128 %0, %1, off"
                 :: "v"((unsigned)(uintptr_t)lds),
                    "v"((unsigned long long)(uintptr_t)gaddr) : "memory");
}
static __device__ __forceinline__ void async_b64(const void* gaddr, const void* lds) {
    asm volatile("global_load_async_to_lds_b64 %0, %1, off"
                 :: "v"((unsigned)(uintptr_t)lds),
                    "v"((unsigned long long)(uintptr_t)gaddr) : "memory");
}

// ---------------------------------------------------------------------------
// Kernel 0: elementwise fp32 -> bf16 (inputs and weights), packed cvt
// ---------------------------------------------------------------------------
__global__ void __launch_bounds__(256) cvt_kernel(const float* __restrict__ src,
                                                  unsigned short* __restrict__ dst,
                                                  int npairs) {
    const int i = blockIdx.x * 256 + threadIdx.x;
    if (i < npairs) {
        v2f f = *(const v2f*)(src + 2 * (size_t)i);
        *(unsigned int*)(dst + 2 * (size_t)i) = pack2bf(f.x, f.y);
    }
}

// ---------------------------------------------------------------------------
// Kernel 1: Y = X @ W^T + b   (X,W already bf16 -> Y bf16)
// workgroup computes a 128x64 tile; k-slabs (X:128x32, W:64x32) streamed into
// double-buffered LDS with global_load_async_to_lds_b128; 3 async ops per
// thread per stage; s_wait_asynccnt 3 overlaps stage t+1 with compute of t.
// Each of the 8 waves does 4 WMMAs per 32-deep k-step.
// ---------------------------------------------------------------------------
__global__ void __launch_bounds__(256) proj_kernel(const unsigned short* __restrict__ X,
                                                   const unsigned short* __restrict__ W,
                                                   const float* __restrict__ bias,
                                                   unsigned short* __restrict__ Y) {
    __shared__ unsigned short Xs[2][128 * 32];   // 8 KB per buffer
    __shared__ unsigned short Ws[2][64 * 32];    // 4 KB per buffer

    const int tid  = threadIdx.x;
    const int wid  = tid >> 5;
    const int lane = tid & 31;
    const int tm   = blockIdx.x >> 4;       // 32 row bands of 128
    const int tn   = blockIdx.x & 15;       // 16 col bands of 64
    const int half = lane >> 4;
    const int lm   = lane & 15;
    const int rowl = wid * 16 + lm;         // this wave's A-frag row in the tile

    const int NK = D_ / 32;                 // 32 k-steps

    // staging geometry: X slab = 512 b128 chunks (2/thread), W slab = 256 (1/thread)
    const int xrow0 = tid >> 2,         xseg0 = tid & 3;
    const int xrow1 = (tid + 256) >> 2, xseg1 = (tid + 256) & 3;
    const int wrow  = tid >> 2,         wseg  = tid & 3;

    auto stage = [&](int buf, int k0) {
        async_b128(X + (size_t)(tm * 128 + xrow0) * D_ + k0 + xseg0 * 8,
                   &Xs[buf][xrow0 * 32 + xseg0 * 8]);
        async_b128(X + (size_t)(tm * 128 + xrow1) * D_ + k0 + xseg1 * 8,
                   &Xs[buf][xrow1 * 32 + xseg1 * 8]);
        async_b128(W + (size_t)(tn * 64 + wrow) * D_ + k0 + wseg * 8,
                   &Ws[buf][wrow * 32 + wseg * 8]);
    };

    v8f acc[4] = {v8f{}, v8f{}, v8f{}, v8f{}};

    stage(0, 0);
    for (int kt = 0; kt < NK; ++kt) {
        __syncthreads();                       // readers of the buffer we re-fill are done
        if (kt + 1 < NK) {
            stage((kt + 1) & 1, (kt + 1) * 32);
            asm volatile("s_wait_asynccnt 0x3" ::: "memory");   // stage kt landed
        } else {
            asm volatile("s_wait_asynccnt 0x0" ::: "memory");
        }
        __syncthreads();                       // stage kt visible to all waves

        const unsigned short* xb = Xs[kt & 1];
        const unsigned short* wb = Ws[kt & 1];
        AF a;
#pragma unroll
        for (int i = 0; i < 4; ++i) {
            a.u[i]     = *(const unsigned int*)&xb[rowl * 32 + 16 * half + 2 * i];
            a.u[4 + i] = *(const unsigned int*)&xb[rowl * 32 + 8 + 16 * half + 2 * i];
        }
        AF bm[4];
#pragma unroll
        for (int ns = 0; ns < 4; ++ns)
#pragma unroll
            for (int i = 0; i < 8; ++i)
                bm[ns].u[i] = *(const unsigned int*)&wb[(ns * 16 + lm) * 32 + 16 * half + 2 * i];
#pragma unroll
        for (int ns = 0; ns < 4; ++ns)
            acc[ns] = WMMA_BF16(a.v, bm[ns].v, acc[ns]);
    }

#pragma unroll
    for (int ns = 0; ns < 4; ++ns) {
        const int n  = tn * 64 + ns * 16 + lm;
        const float bn = bias[n];
#pragma unroll
        for (int r = 0; r < 8; ++r) {
            const int m = tm * 128 + wid * 16 + r + 8 * half;   // D-matrix row layout
            Y[(size_t)m * D_ + n] = f2bf(acc[ns][r] + bn);
        }
    }
}

// ---------------------------------------------------------------------------
// Kernel 2: streaming M = rowmax(QK^T) - rowmean(QK^T) per (b,h,q)
// 8 waves/block share (b,h); K tiles (16x64 bf16, 2KB) double-buffered via
// global_load_async_to_lds_b64 (1 op/thread/stage, s_wait_asynccnt 1).
// ---------------------------------------------------------------------------
__global__ void __launch_bounds__(256) scores_stats_kernel(const unsigned short* __restrict__ Qb,
                                                           const unsigned short* __restrict__ Kb,
                                                           float* __restrict__ Mval) {
    __shared__ unsigned short Ks[2][16 * 64];     // [kcol][d], 2 KB per buffer

    const int tid  = threadIdx.x;
    const int wid  = tid >> 5;
    const int lane = tid & 31;
    const int bh   = blockIdx.x >> 4;              // 0..31
    const int qt   = (blockIdx.x & 15) * 8 + wid;  // 0..127
    const int b    = bh >> 4, h = bh & 15;
    const int half = lane >> 4, lm = lane & 15;

    // staging assignment: 256 threads x 8 bytes = whole 2KB tile
    const int scol = tid >> 4;                     // 0..15 key column
    const int sseg = tid & 15;                     // 16 x 4-short segments
    const unsigned short* Kbase = Kb + (size_t)b * L_ * D_ + h * DK_ + sseg * 4;

    auto stage = [&](int buf, int kt) {
        async_b64(Kbase + (size_t)(kt * 16 + scol) * D_, &Ks[buf][scol * 64 + sseg * 4]);
    };

    const unsigned short* Qrow = Qb + ((size_t)b * L_ + qt * 16 + lm) * D_ + h * DK_;
    AF aq0, aq1;
#pragma unroll
    for (int i = 0; i < 4; ++i) {
        aq0.u[i]     = *(const unsigned int*)(Qrow + 16 * half + 2 * i);
        aq0.u[4 + i] = *(const unsigned int*)(Qrow + 8 + 16 * half + 2 * i);
        aq1.u[i]     = *(const unsigned int*)(Qrow + 32 + 16 * half + 2 * i);
        aq1.u[4 + i] = *(const unsigned int*)(Qrow + 40 + 16 * half + 2 * i);
    }

    float rmax[8], rsum[8];
#pragma unroll
    for (int r = 0; r < 8; ++r) { rmax[r] = -INFINITY; rsum[r] = 0.0f; }

    const int NT = L_ / 16;
    stage(0, 0);
    for (int kt = 0; kt < NT; ++kt) {
        __syncthreads();
        if (kt + 1 < NT) {
            stage((kt + 1) & 1, kt + 1);
            asm volatile("s_wait_asynccnt 0x1" ::: "memory");
        } else {
            asm volatile("s_wait_asynccnt 0x0" ::: "memory");
        }
        __syncthreads();

        const unsigned short* kb = Ks[kt & 1];
        AF bk0, bk1;
#pragma unroll
        for (int i = 0; i < 8; ++i) {
            bk0.u[i] = *(const unsigned int*)&kb[lm * 64 + 16 * half + 2 * i];
            bk1.u[i] = *(const unsigned int*)&kb[lm * 64 + 32 + 16 * half + 2 * i];
        }
        v8f acc = {};
        acc = WMMA_BF16(aq0.v, bk0.v, acc);
        acc = WMMA_BF16(aq1.v, bk1.v, acc);
#pragma unroll
        for (int r = 0; r < 8; ++r) {
            const float x = acc[r];
            rmax[r] = fmaxf(rmax[r], x);
            rsum[r] += x;
        }
    }
#pragma unroll
    for (int r = 0; r < 8; ++r) {
        float mx = rmax[r], sm = rsum[r];
#pragma unroll
        for (int m = 1; m < 16; m <<= 1) {
            mx = fmaxf(mx, __shfl_xor(mx, m, 32));
            sm += __shfl_xor(sm, m, 32);
        }
        if (lm == 0)
            Mval[(size_t)bh * L_ + qt * 16 + 8 * half + r] = mx - sm * (1.0f / (float)L_);
    }
}

// ---------------------------------------------------------------------------
// Kernel 3: top-U selection per (b,h); jax.lax.top_k order (desc, ties->low idx)
// ---------------------------------------------------------------------------
__global__ void __launch_bounds__(256) topk_kernel(const float* __restrict__ Mval,
                                                   int* __restrict__ Mtop) {
    const int bh  = blockIdx.x;
    const int tid = threadIdx.x;
    __shared__ float sv[L_];
    __shared__ float rv[256];
    __shared__ int   ri[256];

    for (int i = tid; i < L_; i += 256) sv[i] = Mval[(size_t)bh * L_ + i];
    __syncthreads();

    for (int u = 0; u < U_; ++u) {
        float best = -INFINITY; int bi = L_;
        for (int i = tid; i < L_; i += 256) {
            const float x = sv[i];
            if (x > best || (x == best && i < bi)) { best = x; bi = i; }
        }
        rv[tid] = best; ri[tid] = bi;
        __syncthreads();
        for (int s = 128; s > 0; s >>= 1) {
            if (tid < s) {
                const float xv = rv[tid + s]; const int xi = ri[tid + s];
                if (xv > rv[tid] || (xv == rv[tid] && xi < ri[tid])) { rv[tid] = xv; ri[tid] = xi; }
            }
            __syncthreads();
        }
        if (tid == 0) { Mtop[bh * UP_ + u] = ri[0]; sv[ri[0]] = -INFINITY; }
        __syncthreads();
    }
    if (tid < UP_ - U_) Mtop[bh * UP_ + U_ + tid] = 0;   // pad rows
}

// ---------------------------------------------------------------------------
// Kernel 4: selected-query scores (scaled, causal-masked) + per-row max
// ---------------------------------------------------------------------------
__global__ void __launch_bounds__(256) sel_scores_kernel(const unsigned short* __restrict__ Qb,
                                                         const unsigned short* __restrict__ Kb,
                                                         const int* __restrict__ Mtop,
                                                         float* __restrict__ scores,
                                                         float* __restrict__ rowmax) {
    const int wid  = threadIdx.x >> 5;
    const int lane = threadIdx.x & 31;
    const int g    = blockIdx.x * 8 + wid;        // 96 waves
    const int bh   = g / 3, qt = g % 3;
    const int b    = bh >> 4, h = bh & 15;
    const int half = lane >> 4, lm = lane & 15;

    const int qidxA = Mtop[bh * UP_ + qt * 16 + lm];       // gathered A rows
    const unsigned short* Qrow = Qb + ((size_t)b * L_ + qidxA) * D_ + h * DK_;
    AF aq0, aq1;
#pragma unroll
    for (int i = 0; i < 4; ++i) {
        aq0.u[i]     = *(const unsigned int*)(Qrow + 16 * half + 2 * i);
        aq0.u[4 + i] = *(const unsigned int*)(Qrow + 8 + 16 * half + 2 * i);
        aq1.u[i]     = *(const unsigned int*)(Qrow + 32 + 16 * half + 2 * i);
        aq1.u[4 + i] = *(const unsigned int*)(Qrow + 40 + 16 * half + 2 * i);
    }
    int qpos[8];
#pragma unroll
    for (int r = 0; r < 8; ++r) qpos[r] = Mtop[bh * UP_ + qt * 16 + 8 * half + r];

    float rmax[8];
#pragma unroll
    for (int r = 0; r < 8; ++r) rmax[r] = -INFINITY;

    for (int kt = 0; kt < L_ / 16; ++kt) {
        const int kcol = kt * 16 + lm;
        const unsigned short* Krow = Kb + ((size_t)b * L_ + kcol) * D_ + h * DK_;
        AF bk0, bk1;
#pragma unroll
        for (int i = 0; i < 8; ++i) {
            bk0.u[i] = *(const unsigned int*)(Krow + 16 * half + 2 * i);
            bk1.u[i] = *(const unsigned int*)(Krow + 32 + 16 * half + 2 * i);
        }
        v8f acc = {};
        acc = WMMA_BF16(aq0.v, bk0.v, acc);
        acc = WMMA_BF16(aq1.v, bk1.v, acc);
#pragma unroll
        for (int r = 0; r < 8; ++r) {
            float s = acc[r] * 0.125f;                     // 1/sqrt(DK)
            if (kcol > qpos[r]) s += -1.0e9f;              // gathered causal row
            const int u = qt * 16 + 8 * half + r;
            if (u < U_) scores[((size_t)bh * U_ + u) * L_ + kcol] = s;
            rmax[r] = fmaxf(rmax[r], s);
        }
    }
#pragma unroll
    for (int r = 0; r < 8; ++r) {
        float mx = rmax[r];
#pragma unroll
        for (int m = 1; m < 16; m <<= 1) mx = fmaxf(mx, __shfl_xor(mx, m, 32));
        if (lm == 0) rowmax[bh * UP_ + qt * 16 + 8 * half + r] = mx;
    }
}

// ---------------------------------------------------------------------------
// Kernel 5: softmax rows in place (fp32) + bf16 mirror for final WMMA
// ---------------------------------------------------------------------------
__global__ void __launch_bounds__(256) softmax_kernel(float* __restrict__ scores,
                                                      const float* __restrict__ rowmax,
                                                      unsigned short* __restrict__ probsb) {
    const int wid  = threadIdx.x >> 5;
    const int lane = threadIdx.x & 31;
    const int g    = blockIdx.x * 8 + wid;        // 1280 waves
    const int bh   = g / U_, u = g % U_;

    float* row = scores + ((size_t)bh * U_ + u) * L_;
    const float mx = rowmax[bh * UP_ + u];

    float sum = 0.0f;
    for (int i = lane; i < L_; i += 32) sum += __expf(row[i] - mx);
#pragma unroll
    for (int m = 1; m < 32; m <<= 1) sum += __shfl_xor(sum, m, 32);
    const float inv = 1.0f / sum;

    unsigned short* brow = probsb + ((size_t)bh * UP_ + u) * L_;
    for (int i = lane; i < L_; i += 32) {
        const float p = __expf(row[i] - mx) * inv;
        row[i]  = p;
        brow[i] = f2bf(p);
    }
    if (u < UP_ - U_) {                            // zero the pad rows once
        unsigned short* prow = probsb + ((size_t)bh * UP_ + U_ + u) * L_;
        for (int i = lane; i < L_; i += 32) prow[i] = 0;
    }
}

// ---------------------------------------------------------------------------
// Kernel 6: attn[b,u, h*DK+dk] = probs[bh,u,:] @ V[bh,:,dk]   (bf16 WMMA)
// ---------------------------------------------------------------------------
__global__ void __launch_bounds__(256) attn_kernel(const unsigned short* __restrict__ probsb,
                                                   const unsigned short* __restrict__ Vb,
                                                   float* __restrict__ out) {
    const int wid  = threadIdx.x >> 5;
    const int lane = threadIdx.x & 31;
    const int g    = blockIdx.x * 8 + wid;        // 384 waves
    const int bh   = g / 12;
    const int rem  = g % 12;
    const int qt   = rem >> 2, nt = rem & 3;
    const int b    = bh >> 4, h = bh & 15;
    const int half = lane >> 4, lm = lane & 15;

    const unsigned short* Arow = probsb + ((size_t)bh * UP_ + qt * 16 + lm) * L_;
    const unsigned short* Vcol = Vb + (size_t)b * L_ * D_ + h * DK_ + nt * 16 + lm;

    v8f acc = {};
    for (int kc = 0; kc < L_ / 32; ++kc) {
        const int kbase = kc * 32;
        AF a, bm;
#pragma unroll
        for (int i = 0; i < 4; ++i) {
            a.u[i]     = *(const unsigned int*)(Arow + kbase + 16 * half + 2 * i);
            a.u[4 + i] = *(const unsigned int*)(Arow + kbase + 8 + 16 * half + 2 * i);
        }
#pragma unroll
        for (int i = 0; i < 8; ++i) {
            const int k0 = kbase + 16 * half + 2 * i;      // strided V rows
            bm.s[2 * i]     = Vcol[(size_t)k0 * D_];
            bm.s[2 * i + 1] = Vcol[(size_t)(k0 + 1) * D_];
        }
        acc = WMMA_BF16(a.v, bm.v, acc);
    }
#pragma unroll
    for (int r = 0; r < 8; ++r) {
        const int u = qt * 16 + r + 8 * half;
        if (u < U_)
            out[((size_t)b * U_ + u) * D_ + h * DK_ + nt * 16 + lm] = acc[r];
    }
}

// ---------------------------------------------------------------------------
extern "C" void kernel_launch(void* const* d_in, const int* in_sizes, int n_in,
                              void* d_out, int out_size, void* d_ws, size_t ws_size,
                              hipStream_t stream) {
    (void)in_sizes; (void)n_in; (void)out_size; (void)ws_size;

    const float* query = (const float*)d_in[0];
    const float* key   = (const float*)d_in[1];
    const float* value = (const float*)d_in[2];
    const float* Wq    = (const float*)d_in[3];
    const float* bq    = (const float*)d_in[4];
    const float* Wk    = (const float*)d_in[5];
    const float* bk    = (const float*)d_in[6];
    const float* Wv    = (const float*)d_in[7];
    const float* bv    = (const float*)d_in[8];

    char* ws = (char*)d_ws;
    const size_t xElems = (size_t)B_ * L_ * D_;            // 4M
    const size_t wElems = (size_t)D_ * D_;                 // 1M
    size_t off = 0;
    unsigned short* Xqb = (unsigned short*)(ws + off); off += xElems * 2;
    unsigned short* Xkb = (unsigned short*)(ws + off); off += xElems * 2;
    unsigned short* Xvb = (unsigned short*)(ws + off); off += xElems * 2;
    unsigned short* Wqb = (unsigned short*)(ws + off); off += wElems * 2;
    unsigned short* Wkb = (unsigned short*)(ws + off); off += wElems * 2;
    unsigned short* Wvb = (unsigned short*)(ws + off); off += wElems * 2;
    unsigned short* Qb  = (unsigned short*)(ws + off); off += xElems * 2;
    unsigned short* Kb  = (unsigned short*)(ws + off); off += xElems * 2;
    unsigned short* Vb  = (unsigned short*)(ws + off); off += xElems * 2;
    float*          Mval   = (float*)(ws + off); off += (size_t)B_ * H_ * L_ * 4;
    int*            Mtop   = (int*)(ws + off);   off += (size_t)B_ * H_ * UP_ * 4;
    float*          rowmax = (float*)(ws + off); off += (size_t)B_ * H_ * UP_ * 4;
    unsigned short* probsb = (unsigned short*)(ws + off);

    float* attn  = (float*)d_out;                          // [B, U, D]
    float* probs = attn + (size_t)B_ * U_ * D_;            // [B, H, U, L]

    // 0) fp32 -> bf16 for GEMM operands (packed v_cvt_pk_bf16_f32)
    cvt_kernel<<<(int)(xElems / 2 / 256), 256, 0, stream>>>(query, Xqb, (int)(xElems / 2));
    cvt_kernel<<<(int)(xElems / 2 / 256), 256, 0, stream>>>(key,   Xkb, (int)(xElems / 2));
    cvt_kernel<<<(int)(xElems / 2 / 256), 256, 0, stream>>>(value, Xvb, (int)(xElems / 2));
    cvt_kernel<<<(int)(wElems / 2 / 256), 256, 0, stream>>>(Wq,    Wqb, (int)(wElems / 2));
    cvt_kernel<<<(int)(wElems / 2 / 256), 256, 0, stream>>>(Wk,    Wkb, (int)(wElems / 2));
    cvt_kernel<<<(int)(wElems / 2 / 256), 256, 0, stream>>>(Wv,    Wvb, (int)(wElems / 2));

    // 1) projections: async double-buffered LDS GEMM, bias fused
    proj_kernel<<<512, 256, 0, stream>>>(Xqb, Wqb, bq, Qb);
    proj_kernel<<<512, 256, 0, stream>>>(Xkb, Wkb, bk, Kb);
    proj_kernel<<<512, 256, 0, stream>>>(Xvb, Wvb, bv, Vb);

    // 2) streaming sparsity measure M = rowmax - rowmean of QK^T
    scores_stats_kernel<<<512, 256, 0, stream>>>(Qb, Kb, Mval);

    // 3) top-U query selection per head
    topk_kernel<<<B_ * H_, 256, 0, stream>>>(Mval, Mtop);

    // 4) gathered-query scores, scaled + causal-masked, row-max tracked
    sel_scores_kernel<<<12, 256, 0, stream>>>(Qb, Kb, Mtop, probs, rowmax);

    // 5) softmax in place + bf16 mirror
    softmax_kernel<<<160, 256, 0, stream>>>(probs, rowmax, probsb);

    // 6) attn = probs @ V
    attn_kernel<<<48, 256, 0, stream>>>(probsb, Vb, attn);
}
